// MethodGNN_36601711296793
// MI455X (gfx1250) — matrix operand order repaired and checked by reading, not compiled, not run on hardware
//
#include <hip/hip_runtime.h>

// ---------------------------------------------------------------------------
// 2-layer GCN for MI455X (gfx1250, wave32).
//   h1 = relu(Ahat @ (x @ W1) + b1); out = log_softmax(Ahat @ (h1 @ W2) + b2)
// GEMMs: bf16 WMMA (v_wmma_f32_16x16x32_bf16, f32 accum), LDS-staged tiles,
// branch-free staging (address clamping + select-zero on the peeled K-tail).
// Sparse aggregation: self-loop term fused into buffer init, edge-parallel
// gather + non-returning f32 atomic scatter (the HBM-roofline-bound phase).
// ---------------------------------------------------------------------------

#define F_IN      1433
#define F_HID     789
#define F_OUT     7
#define F_OUT_PAD 16

typedef __bf16 bf16_t;
typedef __attribute__((ext_vector_type(16))) bf16_t v16bf;
typedef __attribute__((ext_vector_type(8)))  float  v8f;

union Frag16 { v16bf v; unsigned int u[8]; };

__device__ __forceinline__ unsigned short f2bf(float f) {
    // round-to-nearest-even f32 -> bf16
    unsigned int u = __float_as_uint(f);
    u += 0x7FFFu + ((u >> 16) & 1u);
    return (unsigned short)(u >> 16);
}

// ------------------------------ utility ------------------------------------

__global__ void fill_f32(float* __restrict__ p, float v, long long n) {
    long long i      = (long long)blockIdx.x * blockDim.x + threadIdx.x;
    long long stride = (long long)gridDim.x * blockDim.x;
    for (; i < n; i += stride) p[i] = v;
}

__global__ void degree_count(const int* __restrict__ dst, float* __restrict__ deg, int E) {
    int i = blockIdx.x * blockDim.x + threadIdx.x;
    if (i < E) atomicAdd(&deg[dst[i]], 1.0f);
}

__global__ void rsqrt_inplace(float* __restrict__ p, int n) {
    int i = blockIdx.x * blockDim.x + threadIdx.x;
    if (i < n) p[i] = rsqrtf(p[i]);   // deg >= 1 (self loop), always valid
}

// ------------------------------ GEMM 1 -------------------------------------
// H[M x 789] = X[M x 1433] @ W1[1433 x 789], bf16 WMMA, f32 accum.
// Block: 256 thr = 8 waves (4 M-waves x 2 N-waves), tile 128x64, BK=32.

#define G_BM 128
#define G_BN 64
#define G_BK 32
#define LDK  (G_BK + 8)   // ushort row stride: 80B rows, 16B aligned, bank-skewed

__global__ __launch_bounds__(256) void gemm_x_w1(
    const float* __restrict__ X, const float* __restrict__ W,
    float* __restrict__ H, int M)
{
    const int K = F_IN, N = F_HID;
    const int K_FULL = K & ~(G_BK - 1);          // 1408
    __shared__ unsigned short As[G_BM][LDK];
    __shared__ unsigned short Bs[G_BN][LDK];     // staged K-transposed: [n][k]

    const int tid  = threadIdx.x;
    const int lane = tid & 31;
    const int wid  = tid >> 5;
    const int wm   = wid & 3;        // 32-row slab
    const int wn   = wid >> 2;       // 32-col slab
    const int rowBase = blockIdx.x * G_BM;
    const int colBase = blockIdx.y * G_BN;

    v8f acc[2][2] = {};

    // per-lane fragment geometry (CDNA5 wave32 16-bit WMMA layouts)
    const int a_m  = lane & 15;
    const int a_kb = (lane >> 4) << 3;   // A: lanes 16-31 start at K=8
    const int b_n  = lane & 15;
    const int b_kb = (lane >> 4) << 4;   // unused; B is read with a_kb, same k order as A

    // staging geometry: lanes sweep contiguous global addresses
    const int ac  = tid & 31;            // A: k-within-tile (contig per row)
    const int ar0 = (tid >> 5) << 4;     // A: 16-row slab per wave
    const int bnn = tid & 63;            // B: n (contig within W row)
    const int bk0 = tid >> 6;            // B: k start 0..3 (step 4)
    const int gcolC = min(colBase + bnn, N - 1);   // clamped, never stored OOB

    auto compute_tile = [&]() {
        __syncthreads();
        Frag16 a[2], b[2];
        #pragma unroll
        for (int mt = 0; mt < 2; ++mt) {
            const int r = (wm << 5) + (mt << 4) + a_m;
            #pragma unroll
            for (int p = 0; p < 8; ++p) {
                const int kk = a_kb + ((p < 4) ? (p << 1) : (8 + (p << 1)));
                a[mt].u[p] = *(const unsigned int*)&As[r][kk];
            }
        }
        #pragma unroll
        for (int nt = 0; nt < 2; ++nt) {
            const int n = (wn << 5) + (nt << 4) + b_n;
            #pragma unroll
            for (int p = 0; p < 8; ++p)
                b[nt].u[p] = *(const unsigned int*)&Bs[n][a_kb + ((p < 4) ? (p << 1) : (8 + (p << 1)))];
        }
        #pragma unroll
        for (int mt = 0; mt < 2; ++mt)
            #pragma unroll
            for (int nt = 0; nt < 2; ++nt)
                acc[mt][nt] = __builtin_amdgcn_wmma_f32_16x16x32_bf16(
                    false, a[mt].v, false, b[nt].v,
                    (short)0, acc[mt][nt], false, false);
        __syncthreads();
    };

    // ---- main loop: full K-tiles, branch-free staging ----
    for (int k0 = 0; k0 < K_FULL; k0 += G_BK) {
        #pragma unroll
        for (int j = 0; j < 16; ++j) {
            const int r = ar0 + j;
            const long long gr = min(rowBase + r, M - 1);     // clamp, not branch
            As[r][ac] = f2bf(X[gr * K + k0 + ac]);
        }
        __builtin_prefetch(&X[(long long)min(rowBase + ar0, M - 1) * K + k0 + G_BK + ac], 0, 0);
        #pragma unroll
        for (int j = 0; j < 8; ++j) {
            const int kk = bk0 + (j << 2);
            Bs[bnn][kk] = f2bf(W[(long long)(k0 + kk) * N + gcolC]);
        }
        compute_tile();
    }

    // ---- peeled K-tail (25 of 32 valid): clamp + select-zero, branch-free ----
    {
        #pragma unroll
        for (int j = 0; j < 16; ++j) {
            const int r = ar0 + j;
            const long long gr = min(rowBase + r, M - 1);
            const int gc = K_FULL + ac;
            const float v = X[gr * K + min(gc, K - 1)];
            As[r][ac] = f2bf(gc < K ? v : 0.0f);
        }
        #pragma unroll
        for (int j = 0; j < 8; ++j) {
            const int kk = bk0 + (j << 2);
            const int gk = K_FULL + kk;
            const float v = W[(long long)min(gk, K - 1) * N + gcolC];
            Bs[bnn][kk] = f2bf(gk < K ? v : 0.0f);
        }
        compute_tile();
    }

    // ---- store C (guard hoisted: interior tiles store unpredicated) ----
    const int c_n  = lane & 15;
    const int c_r8 = (lane >> 4) << 3;
    if (rowBase + G_BM <= M && colBase + G_BN <= N) {
        #pragma unroll
        for (int mt = 0; mt < 2; ++mt)
            #pragma unroll
            for (int nt = 0; nt < 2; ++nt) {
                const int gc = colBase + (wn << 5) + (nt << 4) + c_n;
                #pragma unroll
                for (int r = 0; r < 8; ++r) {
                    const int gr = rowBase + (wm << 5) + (mt << 4) + c_r8 + r;
                    H[(long long)gr * N + gc] = acc[mt][nt][r];
                }
            }
    } else {
        #pragma unroll
        for (int mt = 0; mt < 2; ++mt)
            #pragma unroll
            for (int nt = 0; nt < 2; ++nt) {
                const int gc = colBase + (wn << 5) + (nt << 4) + c_n;
                #pragma unroll
                for (int r = 0; r < 8; ++r) {
                    const int gr = rowBase + (wm << 5) + (mt << 4) + c_r8 + r;
                    if (gr < M && gc < N) H[(long long)gr * N + gc] = acc[mt][nt][r];
                }
            }
    }
}

// ------------------------------ GEMM 2 -------------------------------------
// H2[M x 16pad] = relu(A + b1) @ W2[789 x 7 padded to 16]. 8 waves, 16 rows each.

__global__ __launch_bounds__(256) void gemm_h_w2(
    const float* __restrict__ A, const float* __restrict__ b1,
    const float* __restrict__ W, float* __restrict__ H2, int M)
{
    const int K = F_HID;
    const int K_FULL = K & ~(G_BK - 1);          // 768
    __shared__ unsigned short As[G_BM][LDK];
    __shared__ unsigned short Bs[16][LDK];

    const int tid  = threadIdx.x;
    const int lane = tid & 31;
    const int wid  = tid >> 5;
    const int rowBase = blockIdx.x * G_BM;

    v8f acc = {};

    const int a_m  = lane & 15;
    const int a_kb = (lane >> 4) << 3;
    const int b_n  = lane & 15;
    const int b_kb = (lane >> 4) << 4;

    const int ac  = tid & 31;            // A: k-within-tile
    const int ar0 = (tid >> 5) << 4;     // A: 16-row slab per wave
    const int n2  = tid & 15;            // B: n 0..15
    const int k2  = tid >> 4;            // B: k 0..15 (plus +16)

    auto compute_tile = [&]() {
        __syncthreads();
        Frag16 a, b;
        const int r = (wid << 4) + a_m;
        #pragma unroll
        for (int p = 0; p < 8; ++p) {
            const int kk = a_kb + ((p < 4) ? (p << 1) : (8 + (p << 1)));
            a.u[p] = *(const unsigned int*)&As[r][kk];
        }
        #pragma unroll
        for (int p = 0; p < 8; ++p)
            b.u[p] = *(const unsigned int*)&Bs[b_n][a_kb + ((p < 4) ? (p << 1) : (8 + (p << 1)))];
        acc = __builtin_amdgcn_wmma_f32_16x16x32_bf16(
            false, a.v, false, b.v, (short)0, acc, false, false);
        __syncthreads();
    };

    // ---- main loop: full K-tiles ----
    for (int k0 = 0; k0 < K_FULL; k0 += G_BK) {
        #pragma unroll
        for (int j = 0; j < 16; ++j) {
            const int r = ar0 + j;
            const long long gr = min(rowBase + r, M - 1);
            const int gc = k0 + ac;
            As[r][ac] = f2bf(fmaxf(A[gr * K + gc] + b1[gc], 0.0f));  // fused +b1, ReLU
        }
        #pragma unroll
        for (int j = 0; j < 2; ++j) {
            const int kk = k2 + (j << 4);
            const int gk = k0 + kk;
            const float v = W[min(gk * F_OUT + n2, F_HID * F_OUT - 1)];
            Bs[n2][kk] = f2bf(n2 < F_OUT ? v : 0.0f);    // pad N 7->16 with zeros
        }
        compute_tile();
    }

    // ---- peeled K-tail (21 of 32 valid) ----
    {
        #pragma unroll
        for (int j = 0; j < 16; ++j) {
            const int r = ar0 + j;
            const long long gr = min(rowBase + r, M - 1);
            const int gc = K_FULL + ac;
            const int gcc = min(gc, K - 1);
            const float v = fmaxf(A[gr * K + gcc] + b1[gcc], 0.0f);
            As[r][ac] = f2bf(gc < K ? v : 0.0f);
        }
        #pragma unroll
        for (int j = 0; j < 2; ++j) {
            const int kk = k2 + (j << 4);
            const int gk = K_FULL + kk;
            const float v = W[min(gk * F_OUT + n2, F_HID * F_OUT - 1)];
            Bs[n2][kk] = f2bf((n2 < F_OUT && gk < K) ? v : 0.0f);
        }
        compute_tile();
    }

    // ---- store (cols padded to 16; junk cols ignored downstream) ----
    const int c_n  = lane & 15;
    const int c_r8 = (lane >> 4) << 3;
    if (rowBase + G_BM <= M) {
        #pragma unroll
        for (int r = 0; r < 8; ++r) {
            const int gr = rowBase + (wid << 4) + c_r8 + r;
            H2[(long long)gr * F_OUT_PAD + c_n] = acc[r];
        }
    } else {
        #pragma unroll
        for (int r = 0; r < 8; ++r) {
            const int gr = rowBase + (wid << 4) + c_r8 + r;
            if (gr < M) H2[(long long)gr * F_OUT_PAD + c_n] = acc[r];
        }
    }
}

// --------------------------- sparse aggregation ----------------------------

// init out with the self-loop term: out[i] = dinv[i]^2 * h[i]
// (replaces a pure zero-fill pass with useful work)
__global__ void selfloop_init_wide(const float* __restrict__ h, const float* __restrict__ dinv,
                                   float* __restrict__ out, int Nn)
{
    const int i = blockIdx.x;
    const float w = dinv[i] * dinv[i];
    const float* hrow = h + (long long)i * F_HID;
    float* orow = out + (long long)i * F_HID;
    for (int f = threadIdx.x; f < F_HID; f += blockDim.x)
        orow[f] = hrow[f] * w;
}

__global__ void selfloop_init_narrow(const float* __restrict__ h, const float* __restrict__ dinv,
                                     float* __restrict__ out, int Nn)
{
    const int i = blockIdx.x * blockDim.x + threadIdx.x;
    if (i >= Nn) return;
    const float w = dinv[i] * dinv[i];
    #pragma unroll
    for (int f = 0; f < F_OUT; ++f)
        out[i * F_OUT + f] = h[i * F_OUT_PAD + f] * w;
}

// layer 1 edges: one block per edge, 789 features striped over lanes
__global__ void scatter_agg_wide(const int* __restrict__ src, const int* __restrict__ dst,
                                 const float* __restrict__ dinv, const float* __restrict__ h,
                                 float* __restrict__ out)
{
    const int e = blockIdx.x;
    const int s = src[e], d = dst[e];
    const float norm = dinv[s] * dinv[d];
    const float* hrow = h + (long long)s * F_HID;
    float* orow = out + (long long)d * F_HID;
    for (int f = threadIdx.x; f < F_HID; f += blockDim.x)
        atomicAdd(&orow[f], hrow[f] * norm);
}

// layer 2 edges: one thread per edge, 7 features
__global__ void scatter_agg_narrow(const int* __restrict__ src, const int* __restrict__ dst,
                                   const float* __restrict__ dinv, const float* __restrict__ h,
                                   float* __restrict__ out, int E)
{
    const int e = blockIdx.x * blockDim.x + threadIdx.x;
    if (e >= E) return;
    const int s = src[e], d = dst[e];
    const float norm = dinv[s] * dinv[d];
    const float* hrow = h + (long long)s * F_OUT_PAD;
    float* orow = out + (long long)d * F_OUT;
    #pragma unroll
    for (int f = 0; f < F_OUT; ++f)
        atomicAdd(&orow[f], hrow[f] * norm);
}

// --------------------------- epilogue --------------------------------------

__global__ void bias_logsoftmax(const float* __restrict__ t, const float* __restrict__ b2,
                                float* __restrict__ out, int Nn)
{
    const int i = blockIdx.x * blockDim.x + threadIdx.x;
    if (i >= Nn) return;
    float v[F_OUT];
    float m = -3.402823466e38f;
    #pragma unroll
    for (int f = 0; f < F_OUT; ++f) { v[f] = t[i * F_OUT + f] + b2[f]; m = fmaxf(m, v[f]); }
    float s = 0.0f;
    #pragma unroll
    for (int f = 0; f < F_OUT; ++f) s += expf(v[f] - m);
    const float l = logf(s);
    #pragma unroll
    for (int f = 0; f < F_OUT; ++f) out[i * F_OUT + f] = v[f] - m - l;
}

// --------------------------- launcher --------------------------------------

extern "C" void kernel_launch(void* const* d_in, const int* in_sizes, int n_in,
                              void* d_out, int out_size, void* d_ws, size_t ws_size,
                              hipStream_t stream)
{
    (void)n_in; (void)out_size; (void)ws_size;

    const float* x  = (const float*)d_in[0];
    const int*   ei = (const int*)d_in[1];
    const float* W1 = (const float*)d_in[2];
    const float* b1 = (const float*)d_in[3];
    const float* W2 = (const float*)d_in[4];
    const float* b2 = (const float*)d_in[5];
    float* out = (float*)d_out;

    const int Nn = in_sizes[0] / F_IN;
    const int E  = in_sizes[1] / 2;
    const int* src = ei;
    const int* dst = ei + E;

    // workspace layout (floats)
    float* h_lin1 = (float*)d_ws;                       // Nn * F_HID
    float* agg1   = h_lin1 + (size_t)Nn * F_HID;        // Nn * F_HID
    float* deg    = agg1   + (size_t)Nn * F_HID;        // Nn  (becomes dinv in place)
    float* h_lin2 = deg    + (size_t)Nn;                // Nn * F_OUT_PAD
    float* otmp   = h_lin2 + (size_t)Nn * F_OUT_PAD;    // Nn * F_OUT

    // degrees (self-loop => init 1.0) -> dinv
    fill_f32<<<1024, 256, 0, stream>>>(deg, 1.0f, (long long)Nn);
    degree_count<<<(E + 255) / 256, 256, 0, stream>>>(dst, deg, E);
    rsqrt_inplace<<<(Nn + 255) / 256, 256, 0, stream>>>(deg, Nn);

    // layer 1 dense: h_lin1 = x @ W1  (bf16 WMMA)
    dim3 g1((Nn + G_BM - 1) / G_BM, (F_HID + G_BN - 1) / G_BN);
    gemm_x_w1<<<g1, 256, 0, stream>>>(x, W1, h_lin1, Nn);

    // layer 1 aggregate: init with self-loop term, then edge scatter
    selfloop_init_wide<<<Nn, 256, 0, stream>>>(h_lin1, deg, agg1, Nn);
    scatter_agg_wide<<<E, 256, 0, stream>>>(src, dst, deg, h_lin1, agg1);

    // layer 2 dense: relu(agg1 + b1) @ W2  (bf16 WMMA, N padded to 16)
    dim3 g2((Nn + G_BM - 1) / G_BM, 1);
    gemm_h_w2<<<g2, 256, 0, stream>>>(agg1, b1, W2, h_lin2, Nn);

    // layer 2 aggregate: self-loop init, then edge scatter
    selfloop_init_narrow<<<(Nn + 255) / 256, 256, 0, stream>>>(h_lin2, deg, otmp, Nn);
    scatter_agg_narrow<<<(E + 255) / 256, 256, 0, stream>>>(src, dst, deg, h_lin2, otmp, E);

    // + b2, log_softmax
    bias_logsoftmax<<<(Nn + 255) / 256, 256, 0, stream>>>(otmp, b2, out, Nn);
}
